// LocationSensitiveAttention2_88759794139534
// MI455X (gfx1250) — matrix operand order repaired and checked by reading, not compiled, not run on hardware
//
#include <hip/hip_runtime.h>
#include <hip/hip_bf16.h>

// ---- problem constants ----
#define BATCH 32
#define TLEN  4096
#define EDIM  128
#define DDIM  1024
#define HDIM  128
#define CCH   32
#define KW    31
#define ODIM  128

#define TILE_T 128   // rows of T per workgroup
#define NWAVE  8     // 256 threads / 32

// Fragment-ordered bf16 blocks: one 16x16x32 operand fragment = 32 lanes x 16 bf16
// (lane L's 16 elements contiguous -> one 32B LDS vector load per fragment).
#define FRAG_ELEMS 512   // 32*16

// ---- workspace layout (bytes), all fragment-ordered bf16 unless noted ----
#define WS_WENC 0                              // 32 frags (4 ks x 8 nt)   32KB
#define WS_WOUT (WS_WENC + 32*FRAG_ELEMS*2)    // 32 frags                 32KB
#define WS_M1   (WS_WOUT + 32*FRAG_ELEMS*2)    // 8 frags (1 ks x 8 nt)     8KB
#define WS_M2   (WS_M1 + 8*FRAG_ELEMS*2)       // 8 frags                   8KB
#define WS_DSB  (WS_M2 + 8*FRAG_ELEMS*2)       // [B][H] f32 dec-proj+b_enc 16KB

typedef __attribute__((ext_vector_type(16))) __bf16 v16bf;
typedef __attribute__((ext_vector_type(8)))  float  v8f;

// f32 -> bf16 via the compiler's native path (v_cvt on gfx1250 if available)
__device__ __forceinline__ __bf16 f2bf(float f) { return (__bf16)f; }
__device__ __forceinline__ unsigned bfb(float f) {
  return (unsigned)__builtin_bit_cast(unsigned short, (__bf16)f);
}

// Native CDNA5 tanh (TRANS32 op). Guarded builtin; inline-asm fallback with the
// required independent-op spacing after a TRANS instruction.
__device__ __forceinline__ float fast_tanh(float x) {
#if __has_builtin(__builtin_amdgcn_tanhf)
  return __builtin_amdgcn_tanhf(x);
#elif __has_builtin(__builtin_amdgcn_tanh_f32)
  return __builtin_amdgcn_tanh_f32(x);
#else
  float r;
  asm volatile("v_tanh_f32 %0, %1\n\tv_nop\n\tv_nop" : "=v"(r) : "v"(x));
  return r;
#endif
}

// A-matrix (16x32 bf16) lane-element -> K map (ISA 7.12.2)
__device__ __forceinline__ int kmapA(int j, int half) {
  return (j < 8) ? (8 * half + j) : (8 * half + j + 8);
}

// ---- CDNA5 async DMA: global -> LDS, 16B per lane, tracked by ASYNCcnt ----
__device__ __forceinline__ void async_b128(void* lds, const void* g) {
  unsigned off = (unsigned)(size_t)lds;                 // low 32 bits = LDS offset
  unsigned long long ga = (unsigned long long)g;
  asm volatile("global_load_async_to_lds_b128 %0, %1, off"
               :: "v"(off), "v"(ga) : "memory");
}
__device__ __forceinline__ void wait_async0() {
  asm volatile("s_wait_asynccnt 0x0" ::: "memory");
}

// ================= prep kernels =================
// Emit weights directly in fragment order:
// index = ((ks*NT + nt)*32 + lane)*16 + j ; half=lane>>4, n=lane&15
//   B-fragment element (lane,j) = B[kb + (j<8 ? 8*half + j : 16 + 8*half + j - 8)][nt*16 + n]

__global__ void lsa_prep_weights(const float* __restrict__ W_enc,
                                 const float* __restrict__ W_out,
                                 const float* __restrict__ W_att,
                                 const float* __restrict__ W_att2,
                                 const float* __restrict__ K1,
                                 const float* __restrict__ K2,
                                 unsigned char* __restrict__ ws) {
  __bf16* wenc = (__bf16*)(ws + WS_WENC);
  __bf16* wout = (__bf16*)(ws + WS_WOUT);
  __bf16* m1   = (__bf16*)(ws + WS_M1);
  __bf16* m2   = (__bf16*)(ws + WS_M2);
  int i = blockIdx.x * blockDim.x + threadIdx.x;
  if (i < 16384) {                           // Wenc^T fragments
    int j = i & 15, lane = (i >> 4) & 31, nt = (i >> 9) & 7, ks = (i >> 12) & 3;
    int half = lane >> 4, n = lane & 15;
    int e = ks * 32 + ((j < 8) ? (8 * half + j) : (8 * half + j + 8));
    int h = nt * 16 + n;
    wenc[i] = f2bf(W_enc[h * EDIM + e]);
  } else if (i < 32768) {                    // Wout^T fragments
    int i2 = i - 16384;
    int j = i2 & 15, lane = (i2 >> 4) & 31, nt = (i2 >> 9) & 7, ks = (i2 >> 12) & 3;
    int half = lane >> 4, n = lane & 15;
    int hs = ks * 32 + ((j < 8) ? (8 * half + j) : (8 * half + j + 8));
    int o = nt * 16 + n;
    wout[i2] = f2bf(W_out[o * HDIM + hs]);
  } else if (i < 36864) {                    // M1 = W_att@K1 fragments (k=31 -> 0)
    int i3 = i - 32768;
    int j = i3 & 15, lane = (i3 >> 4) & 31, nt = (i3 >> 9) & 7;
    int half = lane >> 4, n = lane & 15;
    int k = (j < 8) ? (8 * half + j) : (8 * half + j + 8);
    int h = nt * 16 + n;
    float s = 0.f;
    if (k < KW) {
      #pragma unroll
      for (int c = 0; c < CCH; ++c) s += W_att[h * CCH + c] * K1[c * KW + k];
    }
    m1[i3] = f2bf(s);
  } else if (i < 40960) {                    // M2
    int i4 = i - 36864;
    int j = i4 & 15, lane = (i4 >> 4) & 31, nt = (i4 >> 9) & 7;
    int half = lane >> 4, n = lane & 15;
    int k = (j < 8) ? (8 * half + j) : (8 * half + j + 8);
    int h = nt * 16 + n;
    float s = 0.f;
    if (k < KW) {
      #pragma unroll
      for (int c = 0; c < CCH; ++c) s += W_att2[h * CCH + c] * K2[c * KW + k];
    }
    m2[i4] = f2bf(s);
  }
}

__global__ void lsa_prep_ds(const float* __restrict__ dec,
                            const float* __restrict__ W_dec,
                            const float* __restrict__ b_enc,
                            unsigned char* __restrict__ ws) {
  float* dsb = (float*)(ws + WS_DSB);
  int i = blockIdx.x * blockDim.x + threadIdx.x;   // 4096 = B*H
  int b = i >> 7, h = i & 127;
  const float* dp = dec + (size_t)b * DDIM;
  const float* wp = W_dec + (size_t)h * DDIM;
  float s = 0.f;
  for (int d = 0; d < DDIM; d += 4)
    s += dp[d] * wp[d] + dp[d+1] * wp[d+1] + dp[d+2] * wp[d+2] + dp[d+3] * wp[d+3];
  dsb[i] = s + b_enc[h];
}

// ================= fused main kernel =================

__global__ __launch_bounds__(256) void lsa_main(
    const float* __restrict__ energy,   // [B,T,E]
    const float* __restrict__ att1,     // [B,T]
    const float* __restrict__ att2,     // [B,T]
    const float* __restrict__ b_out,    // [O]
    const unsigned char* __restrict__ ws,
    float* __restrict__ out)            // [B,T,O]
{
  // LDS (all fragment-ordered bf16): A tile (32 frags, reused for tanh) |
  // weight tile (32 frags: Wenc then Wout) | M1 (8) | M2 (8) | f32 att windows
  __shared__ __align__(16) unsigned char smem[32*FRAG_ELEMS*2 + 32*FRAG_ELEMS*2 +
                                              2*8*FRAG_ELEMS*2 + 2*160*4];
  __bf16* sA  = (__bf16*)smem;                                   // 32 KB
  __bf16* sW  = (__bf16*)(smem + 32*FRAG_ELEMS*2);               // 32 KB
  __bf16* sM1 = (__bf16*)(smem + 64*FRAG_ELEMS*2);               //  8 KB
  __bf16* sM2 = sM1 + 8*FRAG_ELEMS;                              //  8 KB
  float*  sX1 = (float*)(sM2 + 8*FRAG_ELEMS);                    // 640 B
  float*  sX2 = sX1 + 160;

  const int tid  = threadIdx.x;
  const int lane = tid & 31;
  const int wave = tid >> 5;
  const int b    = blockIdx.x >> 5;
  const int t0   = (blockIdx.x & 31) * TILE_T;

  // ---- kick off weight DMA (bf16, already fragment-ordered in ws) ----
  {
    const unsigned char* src = ws + WS_WENC;
    unsigned char* dst = (unsigned char*)sW;
    #pragma unroll
    for (int i = 0; i < 8; ++i) {          // 32KB / (256 thr * 16B) = 8
      int o = (tid + i * 256) * 16;
      async_b128(dst + o, src + o);
    }
    const unsigned char* s1 = ws + WS_M1;
    const unsigned char* s2 = ws + WS_M2;
    #pragma unroll
    for (int i = 0; i < 2; ++i) {          // 8KB each
      int o = (tid + i * 256) * 16;
      async_b128((unsigned char*)sM1 + o, s1 + o);
      async_b128((unsigned char*)sM2 + o, s2 + o);
    }
  }

  // ---- energy tile f32 -> bf16, swizzled into fragment order (overlaps DMA) ----
  {
    const float4* g = (const float4*)(energy + ((size_t)b * TLEN + t0) * EDIM);
    #pragma unroll
    for (int i = 0; i < 16; ++i) {
      int idx = tid + i * 256;             // float4 index in tile
      float4 v = g[idx];
      int t_local = idx >> 5;              // 32 float4 per row
      int e0 = (idx & 31) * 4;             // e0 % 4 == 0 -> 4 consecutive j slots
      int rt = t_local >> 4, m = t_local & 15;
      int ks = e0 >> 5, kk = e0 & 31;
      int half = (kk >> 3) & 1;
      int j0 = kk - 8 * half - 8 * (kk >> 4);
      int di = ((rt * 4 + ks) * 32 + half * 16 + m) * 16 + j0;   // elem index
      unsigned p0 = bfb(v.x) | (bfb(v.y) << 16);
      unsigned p1 = bfb(v.z) | (bfb(v.w) << 16);
      unsigned* d = (unsigned*)(sA + di);  // 8B aligned (j0 in {0,4,8,12})
      d[0] = p0; d[1] = p1;
    }
  }
  // attention windows: global t in [t0-15, t0+144), zero-fill OOB
  for (int j = tid; j < 160; j += 256) {
    int g = t0 - 15 + j;
    bool ok = (g >= 0) && (g < TLEN);
    sX1[j] = ok ? att1[(size_t)b * TLEN + g] : 0.f;
    sX2[j] = ok ? att2[(size_t)b * TLEN + g] : 0.f;
  }
  wait_async0();
  __syncthreads();

  const int m    = lane & 15;
  const int half = lane >> 4;
  const int r0   = wave * 16;

  v8f acc[8] = {};

  // ---- GEMM1: pm = A x Wenc^T (K=128, 4 k-steps); 1 LDS vec load per fragment ----
  #pragma unroll
  for (int ks = 0; ks < 4; ++ks) {
    v16bf a = *(const v16bf*)(sA + (wave * 4 + ks) * FRAG_ELEMS + lane * 16);
    #pragma unroll
    for (int nt = 0; nt < 8; ++nt) {
      v16bf bm = *(const v16bf*)(sW + (ks * 8 + nt) * FRAG_ELEMS + lane * 16);
      acc[nt] = __builtin_amdgcn_wmma_f32_16x16x32_bf16(false, a, false, bm,
                                                        (short)0, acc[nt], false, false);
    }
  }

  // ---- conv k-steps: A[m][k] = att[t0+r0+m+k-15], B = M1/M2 (row 31 = 0) ----
  {
    v16bf a;
    const float* xp = sX1 + r0 + m;
    #pragma unroll
    for (int j = 0; j < 16; ++j) a[j] = f2bf(xp[kmapA(j, half)]);
    #pragma unroll
    for (int nt = 0; nt < 8; ++nt) {
      v16bf bm = *(const v16bf*)(sM1 + nt * FRAG_ELEMS + lane * 16);
      acc[nt] = __builtin_amdgcn_wmma_f32_16x16x32_bf16(false, a, false, bm,
                                                        (short)0, acc[nt], false, false);
    }
  }
  {
    v16bf a;
    const float* xp = sX2 + r0 + m;
    #pragma unroll
    for (int j = 0; j < 16; ++j) a[j] = f2bf(xp[kmapA(j, half)]);
    #pragma unroll
    for (int nt = 0; nt < 8; ++nt) {
      v16bf bm = *(const v16bf*)(sM2 + nt * FRAG_ELEMS + lane * 16);
      acc[nt] = __builtin_amdgcn_wmma_f32_16x16x32_bf16(false, a, false, bm,
                                                        (short)0, acc[nt], false, false);
    }
  }

  // ---- + (decoder proj + b_enc), native v_tanh ----
  const float* dsb = (const float*)(ws + WS_DSB) + b * HDIM;
  #pragma unroll
  for (int nt = 0; nt < 8; ++nt) {
    float d = dsb[nt * 16 + m];
    #pragma unroll
    for (int r = 0; r < 8; ++r) acc[nt][r] = fast_tanh(acc[nt][r] + d);
  }

  __syncthreads();   // all waves done with sW (Wenc) -> DMA Wout into it

  {
    const unsigned char* src = ws + WS_WOUT;
    unsigned char* dst = (unsigned char*)sW;
    #pragma unroll
    for (int i = 0; i < 8; ++i) {
      int o = (tid + i * 256) * 16;
      async_b128(dst + o, src + o);
    }
  }
  // scatter tanh (bf16) into this wave's OWN A fragments, fragment-ordered
  // value (row mp = 8*half+r, h = nt*16+m):
  //   ks2 = nt>>1, half' = (m>>3)&1, j = (m&7) + 8*(nt&1), lane' = half'*16+mp
  {
    int halfp = (m >> 3) & 1;
    #pragma unroll
    for (int nt = 0; nt < 8; ++nt) {
      int jj = (m & 7) + 8 * (nt & 1);
      __bf16* base = sA + (wave * 4 + (nt >> 1)) * FRAG_ELEMS + jj;
      #pragma unroll
      for (int r = 0; r < 8; ++r) {
        int mp = 8 * half + r;
        base[(halfp * 16 + mp) * 16] = f2bf(acc[nt][r]);
      }
    }
  }
  wait_async0();
  __syncthreads();

  // ---- GEMM2: erg = tanh x Wout^T (K=128, 4 k-steps) ----
  v8f acc2[8] = {};
  #pragma unroll
  for (int ks = 0; ks < 4; ++ks) {
    v16bf a = *(const v16bf*)(sA + (wave * 4 + ks) * FRAG_ELEMS + lane * 16);
    #pragma unroll
    for (int nt = 0; nt < 8; ++nt) {
      v16bf bm = *(const v16bf*)(sW + (ks * 8 + nt) * FRAG_ELEMS + lane * 16);
      acc2[nt] = __builtin_amdgcn_wmma_f32_16x16x32_bf16(false, a, false, bm,
                                                         (short)0, acc2[nt], false, false);
    }
  }

  // ---- + b_out, store [B,T,O] f32 ----
  float* op = out + ((size_t)b * TLEN + t0 + r0) * ODIM;
  #pragma unroll
  for (int nt = 0; nt < 8; ++nt) {
    float bo = b_out[nt * 16 + m];
    #pragma unroll
    for (int r = 0; r < 8; ++r) {
      op[(size_t)(8 * half + r) * ODIM + nt * 16 + m] = acc2[nt][r] + bo;
    }
  }
}

// ================= launcher =================

extern "C" void kernel_launch(void* const* d_in, const int* in_sizes, int n_in,
                              void* d_out, int out_size, void* d_ws, size_t ws_size,
                              hipStream_t stream) {
  const float* energy = (const float*)d_in[0];
  const float* dec    = (const float*)d_in[2];
  const float* att1   = (const float*)d_in[3];
  const float* att2   = (const float*)d_in[4];
  const float* W_enc  = (const float*)d_in[7];
  const float* b_enc  = (const float*)d_in[8];
  const float* W_dec  = (const float*)d_in[9];
  const float* W_att  = (const float*)d_in[10];
  const float* W_att2 = (const float*)d_in[11];
  const float* K1     = (const float*)d_in[12];
  const float* K2     = (const float*)d_in[13];
  const float* W_out  = (const float*)d_in[14];
  const float* b_out  = (const float*)d_in[15];
  unsigned char* ws   = (unsigned char*)d_ws;
  float* out          = (float*)d_out;

  lsa_prep_weights<<<160, 256, 0, stream>>>(W_enc, W_out, W_att, W_att2, K1, K2, ws);
  lsa_prep_ds<<<16, 256, 0, stream>>>(dec, W_dec, b_enc, ws);
  lsa_main<<<BATCH * (TLEN / TILE_T), 256, 0, stream>>>(energy, att1, att2, b_out, ws, out);
}
